// Head_83210696392850
// MI455X (gfx1250) — matrix operand order repaired and checked by reading, not compiled, not run on hardware
//
#include <hip/hip_runtime.h>
#include <hip/hip_bf16.h>

#define B_  8
#define T_  2048
#define C_  1024
#define HS_ 64

typedef __attribute__((ext_vector_type(16))) _Float16 v16h;
typedef __attribute__((ext_vector_type(8)))  _Float16 v8h;
typedef __attribute__((ext_vector_type(8)))  float    v8f;
typedef __attribute__((ext_vector_type(4)))  float    v4f;

typedef unsigned int u32x4 __attribute__((ext_vector_type(4)));
typedef int          i32x4 __attribute__((ext_vector_type(4)));
typedef int          i32x8 __attribute__((ext_vector_type(8)));

#if defined(__has_builtin)
#  if __has_builtin(__builtin_amdgcn_tensor_load_to_lds)
#    define HAS_TDM 1
#  endif
#endif
#ifndef HAS_TDM
#  define HAS_TDM 0
#endif

// Build an A-matrix fragment (16x32 f16, wave32 layout):
// lane<16 holds row M=lane, halfs {K kb..kb+7, K kb+8..kb+15}, kb=0;
// lanes>=16 same rows with kb=16.
static __device__ __forceinline__ v16h frag_a(const _Float16* lo, const _Float16* hi) {
  v8h l = *(const v8h*)lo;
  v8h h = *(const v8h*)hi;
  v16h r;
#pragma unroll
  for (int i = 0; i < 8; ++i) { r[i] = l[i]; r[8 + i] = h[i]; }
  return r;
}

static __device__ __forceinline__ v8f wmma_f16(v16h a, v16h b, v8f c) {
  return __builtin_amdgcn_wmma_f32_16x16x32_f16(
      /*neg_a=*/false, a, /*neg_b=*/false, b,
      /*c_mod=*/(short)0, c, /*reuse_a=*/false, /*reuse_b=*/false);
}

#if HAS_TDM
// TDM 2D tile load: global (row-major, 2-byte elements, row stride in
// elements) -> LDS with row padding so rows land at our padded LDS strides.
// D# layout per CDNA5 ISA; tensor dims == tile
// dims (tiles never go OOB here).  pad: interval code v -> 2^(v+1) dwords,
// amount code v -> (v+1) dwords.
static __device__ __forceinline__ void tdm_load_tile_2d(
    unsigned lds_off_bytes, const _Float16* gsrc,
    unsigned tile_d0, unsigned tile_d1, unsigned long row_stride_elems,
    unsigned pad_interval, unsigned pad_amount) {
  unsigned long ga = (unsigned long)gsrc;
  u32x4 g0;
  g0[0] = 1u;                                                  // count=1
  g0[1] = lds_off_bytes;                                       // lds_addr
  g0[2] = (unsigned)ga;                                        // global[31:0]
  g0[3] = (unsigned)((ga >> 32) & 0x01ffffffu) | 0x80000000u;  // global[56:32], type=2
  i32x8 g1;
  g1[0] = (int)((1u << 16) |                    // data_size = 2 bytes
                (1u << 20) |                    // pad_enable
                (pad_interval << 22) | (pad_amount << 25));
  g1[1] = (int)(tile_d0 << 16);                               // tensor_dim0 lo16
  g1[2] = (int)((tile_d0 >> 16) | (tile_d1 << 16));           // dim0 hi | dim1 lo
  g1[3] = (int)((tile_d1 >> 16) | (tile_d0 << 16));           // dim1 hi | tile_dim0
  g1[4] = (int)(tile_d1 & 0xffffu);                           // tile_dim1 | tile_dim2=0
  g1[5] = (int)(unsigned)(row_stride_elems & 0xffffffffu);    // dim0_stride lo32
  g1[6] = (int)(unsigned)(row_stride_elems >> 32);            // dim0_stride hi16
  g1[7] = 0;
  i32x4 z = {0, 0, 0, 0};
#  if __clang_major__ >= 23
  i32x8 z8 = {0, 0, 0, 0, 0, 0, 0, 0};
  __builtin_amdgcn_tensor_load_to_lds(g0, g1, z, z, z8, 0);
#  else
  __builtin_amdgcn_tensor_load_to_lds(g0, g1, z, z, 0);
#  endif
}
#endif

// ---------------------------------------------------------------------------
// Kernel 1: QKV projection.  q/k = f16( x @ W ) row-major [B*T, 64],
// v = f16( x @ Wv ) stored TRANSPOSED as V^T [B][64][T] so the attention
// kernel's P@V B-fragments become contiguous loads.  Q folded scale 1/32.
// Block = 128 threads (4 waves), 64-row M tile, K staged 32 wide in LDS.
// Stores go straight from D-layout registers: Q/K as merged b16 pairs
// (half-wave covers 16 consecutive cols), V^T as packed v8h (accumulator
// register index r == consecutive t) -- no output LDS staging needed.
// ---------------------------------------------------------------------------
__global__ __launch_bounds__(128) void qkv_proj_kernel(
    const float* __restrict__ x, const float* __restrict__ Wq,
    const float* __restrict__ Wk, const float* __restrict__ Wv,
    _Float16* __restrict__ qh, _Float16* __restrict__ kh,
    _Float16* __restrict__ vt) {
  __shared__ __align__(16) _Float16 Xs[64][40];  // [m_local][k_local]
  __shared__ __align__(16) _Float16 Wt[64][40];  // [n][k_local]

  const int tid  = threadIdx.x;
  const int w    = tid >> 5;
  const int lane = tid & 31;
  const int ln   = lane & 15;
  const int hi8  = (lane >> 4) << 3;
  const int kb   = (lane >> 4) * 16;  // A-frag K base
  const int kb16 = (lane >> 4) * 16;  // B-frag K base
  const int m0   = blockIdx.x * 64;
  const int mat  = blockIdx.y;

  const float* Wsel = (mat == 0) ? Wq : (mat == 1) ? Wk : Wv;

  v8f acc[4];
#pragma unroll
  for (int c = 0; c < 4; ++c)
    acc[c] = (v8f){0.f, 0.f, 0.f, 0.f, 0.f, 0.f, 0.f, 0.f};

  for (int kk = 0; kk < C_; kk += 32) {
    // stage x chunk (64x32, f32->f16, row-major)
    {
      const int r  = tid >> 1;
      const int cb = (tid & 1) * 16;
      const float* xp = x + (size_t)(m0 + r) * C_ + kk + cb;
      v8h h0, h1;
#pragma unroll
      for (int g = 0; g < 2; ++g) {
        v4f a = *(const v4f*)(xp + 4 * g);
        v4f b = *(const v4f*)(xp + 8 + 4 * g);
#pragma unroll
        for (int i = 0; i < 4; ++i) {
          h0[4 * g + i] = (_Float16)a[i];
          h1[4 * g + i] = (_Float16)b[i];
        }
      }
      *(v8h*)&Xs[r][cb]     = h0;
      *(v8h*)&Xs[r][cb + 8] = h1;
    }
    // stage W chunk transposed: Wt[n][k] = W[kk+k][n]
    {
      const int c  = tid >> 2;
      const int nb = (tid & 3) * 16;
      const float* wp = Wsel + (size_t)(kk + c) * HS_ + nb;
#pragma unroll
      for (int g = 0; g < 4; ++g) {
        v4f v = *(const v4f*)(wp + 4 * g);
#pragma unroll
        for (int i = 0; i < 4; ++i) Wt[nb + 4 * g + i][c] = (_Float16)v[i];
      }
    }
    __syncthreads();

    v16h a = frag_a(&Xs[16 * w + ln][kb], &Xs[16 * w + ln][kb + 8]);
#pragma unroll
    for (int c = 0; c < 4; ++c) {
      v16h b = *(const v16h*)&Wt[16 * c + ln][kb16];
      acc[c] = wmma_f16(a, b, acc[c]);
    }
    __syncthreads();
  }

  if (mat == 2) {
    // V^T: pack r (consecutive t) into v8h, one 16B store per column chunk
    const int bb = m0 / T_, tb = (m0 % T_) + 16 * w + hi8;  // tile in one batch
#pragma unroll
    for (int c = 0; c < 4; ++c) {
      v8h hv;
#pragma unroll
      for (int r = 0; r < 8; ++r) hv[r] = (_Float16)acc[c][r];
      *(v8h*)(vt + ((size_t)bb * HS_ + 16 * c + ln) * T_ + tb) = hv;
    }
  } else {
    const float oscale = (mat == 0) ? 0.03125f : 1.0f;  // fold C^-0.5 into Q
    _Float16* osel = (mat == 0) ? qh : kh;
#pragma unroll
    for (int r = 0; r < 8; ++r) {
      const int row = m0 + 16 * w + r + hi8;
#pragma unroll
      for (int c = 0; c < 4; ++c)
        osel[(size_t)row * HS_ + 16 * c + ln] = (_Float16)(acc[c][r] * oscale);
    }
  }
}

// ---------------------------------------------------------------------------
// Kernel 2: causal flash attention.  Block = 128 threads (4 waves), 64 query
// rows per block, 32 keys per iteration.  K and V^T tiles are DMA'd into LDS
// by the Tensor Data Mover (double-buffered, s_wait_tensorcnt), shared by all
// 4 waves; score/P@V fragments are contiguous LDS reads in WMMA B-layout.
// ---------------------------------------------------------------------------
__global__ __launch_bounds__(128) void attn_kernel(
    const _Float16* __restrict__ qh, const _Float16* __restrict__ kh,
    const _Float16* __restrict__ vt, float* __restrict__ out) {
  __shared__ __align__(16) _Float16 Ks[2][32][72];  // [buf][key][feat], 144B rows
  __shared__ __align__(16) _Float16 Vs[2][64][40];  // [buf][feat][key],  80B rows
  __shared__ __align__(16) _Float16 Ps[4][16][40];  // per-wave probs staging

  const int tid  = threadIdx.x;
  const int w    = tid >> 5;
  const int lane = tid & 31;
  const int ln   = lane & 15;
  const int hi8  = (lane >> 4) << 3;
  const int kb   = (lane >> 4) * 16;
  const int kb16 = (lane >> 4) * 16;

  const int b  = blockIdx.x >> 5;  // T_/64 == 32 q-tiles per batch
  const int q0 = (blockIdx.x & 31) * 64;
  const size_t base  = (size_t)b * T_ * HS_;  // q/k row-major base
  const size_t vbase = (size_t)b * HS_ * T_;  // V^T base
  const int jend = q0 + 64;

  // Q fragments for this wave's 16 rows (pre-scaled by C^-0.5)
  const _Float16* qp = qh + base + (size_t)(q0 + 16 * w + ln) * HS_;
  const v16h aq0 = frag_a(qp + kb,      qp + kb + 8);
  const v16h aq1 = frag_a(qp + 32 + kb, qp + 32 + kb + 8);

  float m_run[8], l_run[8];
  v8f o[4];
#pragma unroll
  for (int r = 0; r < 8; ++r) { m_run[r] = -1e30f; l_run[r] = 0.f; }
#pragma unroll
  for (int c = 0; c < 4; ++c)
    o[c] = (v8f){0.f, 0.f, 0.f, 0.f, 0.f, 0.f, 0.f, 0.f};

  int p = 0;
#if HAS_TDM
  if (w == 0) {  // prologue: DMA tile j=0 into buffer 0
    tdm_load_tile_2d((unsigned)(uintptr_t)&Ks[0][0][0], kh + base, 64, 32, 64, 4, 3);
    tdm_load_tile_2d((unsigned)(uintptr_t)&Vs[0][0][0], vt + vbase, 32, 64, T_, 3, 3);
  }
#endif

  for (int j = 0; j < jend; j += 32) {
#if HAS_TDM
    if (w == 0) {
      if (j + 32 < jend) {  // DMA next tile into the other buffer
        const int jn = j + 32;
        tdm_load_tile_2d((unsigned)(uintptr_t)&Ks[p ^ 1][0][0],
                         kh + base + (size_t)jn * HS_, 64, 32, 64, 4, 3);
        tdm_load_tile_2d((unsigned)(uintptr_t)&Vs[p ^ 1][0][0],
                         vt + vbase + jn, 32, 64, T_, 3, 3);
        __builtin_amdgcn_s_wait_tensorcnt(2);  // oldest pair (this tile) done
      } else {
        __builtin_amdgcn_s_wait_tensorcnt(0);
      }
    }
#else
    {  // cooperative staging fallback (same padded LDS layout)
      const int kr = tid >> 2, kc = (tid & 3) * 16;
      const _Float16* kp = kh + base + (size_t)(j + kr) * HS_ + kc;
      *(v8h*)&Ks[p][kr][kc]     = *(const v8h*)kp;
      *(v8h*)&Ks[p][kr][kc + 8] = *(const v8h*)(kp + 8);
      const int vh2 = tid >> 1, vc = (tid & 1) * 16;
      const _Float16* vp = vt + vbase + (size_t)vh2 * T_ + j + vc;
      *(v8h*)&Vs[p][vh2][vc]     = *(const v8h*)vp;
      *(v8h*)&Vs[p][vh2][vc + 8] = *(const v8h*)(vp + 8);
      if (j + 32 < T_)
        __builtin_prefetch(kh + base + (size_t)(j + 32 + kr) * HS_, 0, 0);
    }
#endif
    __syncthreads();  // tile p visible to all waves

    // scores S = (Q * C^-0.5) @ K^T for 32 keys (two 16x16 D tiles)
    v8f s0 = (v8f){0.f, 0.f, 0.f, 0.f, 0.f, 0.f, 0.f, 0.f};
    v8f s1 = s0;
    s0 = wmma_f16(aq0, *(const v16h*)&Ks[p][ln][kb16],           s0);
    s0 = wmma_f16(aq1, *(const v16h*)&Ks[p][ln][32 + kb16],      s0);
    s1 = wmma_f16(aq0, *(const v16h*)&Ks[p][16 + ln][kb16],      s1);
    s1 = wmma_f16(aq1, *(const v16h*)&Ks[p][16 + ln][32 + kb16], s1);

    // causal mask + online softmax (D layout; each 16-lane half = 1 row group)
#pragma unroll
    for (int r = 0; r < 8; ++r) {
      const int row = q0 + 16 * w + r + hi8;
      if (j + ln > row)      s0[r] = -1e30f;
      if (j + 16 + ln > row) s1[r] = -1e30f;

      float rm = fmaxf(s0[r], s1[r]);
#pragma unroll
      for (int off = 8; off > 0; off >>= 1) rm = fmaxf(rm, __shfl_xor(rm, off));

      const float mo  = m_run[r];
      const float mn  = fmaxf(mo, rm);
      const float asc = __expf(mo - mn);
      const float p0  = __expf(s0[r] - mn);
      const float p1  = __expf(s1[r] - mn);

      float rs = p0 + p1;
#pragma unroll
      for (int off = 8; off > 0; off >>= 1) rs += __shfl_xor(rs, off);

      m_run[r] = mn;
      l_run[r] = l_run[r] * asc + rs;
#pragma unroll
      for (int c = 0; c < 4; ++c) o[c][r] *= asc;

      const int rl = r + hi8;
      Ps[w][rl][ln]      = (_Float16)p0;
      Ps[w][rl][16 + ln] = (_Float16)p1;
    }
    // Ps write->read is same-wave: DS ops are in-order, no barrier needed.

    // O += P @ V  (A = P 16x32 from LDS, B = V^T rows, contiguous)
    {
      const v16h ap = frag_a(&Ps[w][ln][kb], &Ps[w][ln][kb + 8]);
#pragma unroll
      for (int c = 0; c < 4; ++c) {
        const v16h bv = *(const v16h*)&Vs[p][16 * c + ln][kb16];
        o[c] = wmma_f16(ap, bv, o[c]);
      }
    }
    __syncthreads();  // done reading tile p; safe to DMA over it next round
#if HAS_TDM
    p ^= 1;
#endif
  }

  // normalize and store (f32; lanes 0..15 cover 16 consecutive columns)
#pragma unroll
  for (int r = 0; r < 8; ++r) {
    const float inv = 1.0f / l_run[r];
    const int row = q0 + 16 * w + r + hi8;
    float* op = out + base + (size_t)row * HS_ + ln;
#pragma unroll
    for (int c = 0; c < 4; ++c) op[16 * c] = o[c][r] * inv;
  }
}

// ---------------------------------------------------------------------------
extern "C" void kernel_launch(void* const* d_in, const int* in_sizes, int n_in,
                              void* d_out, int out_size, void* d_ws,
                              size_t ws_size, hipStream_t stream) {
  const float* x  = (const float*)d_in[0];
  const float* Wq = (const float*)d_in[1];
  const float* Wk = (const float*)d_in[2];
  const float* Wv = (const float*)d_in[3];
  float* out = (float*)d_out;

  const size_t N = (size_t)B_ * T_ * HS_;  // 1M elems per tensor
  _Float16* qh = (_Float16*)d_ws;          // Q  f16 [B*T, 64]      (2 MB)
  _Float16* kh = qh + N;                   // K  f16 [B*T, 64]      (2 MB)
  _Float16* vt = kh + N;                   // V^T f16 [B, 64, T]    (2 MB)

  dim3 g1((B_ * T_) / 64, 3);
  qkv_proj_kernel<<<g1, 128, 0, stream>>>(x, Wq, Wk, Wv, qh, kh, vt);

  dim3 g2(B_ * (T_ / 64));
  attn_kernel<<<g2, 128, 0, stream>>>(qh, kh, vt, out);
}
